// BiMamba2Layer_48790828482862
// MI455X (gfx1250) — matrix operand order repaired and checked by reading, not compiled, not run on hardware
//
#include <hip/hip_runtime.h>

// ---------------- types ----------------
typedef __attribute__((ext_vector_type(16))) __bf16       v16bf;
typedef __attribute__((ext_vector_type(8)))  __bf16       v8bf;
typedef __attribute__((ext_vector_type(8)))  float        v8f;
typedef __attribute__((ext_vector_type(4)))  unsigned int u32x4;
typedef __attribute__((ext_vector_type(8)))  unsigned int u32x8;

#define MROWS   15360LL      // B*L = 16*960
#define LSEQ    960
#define DMODEL  512
#define DINNER  1024
#define DSTATE  16
#define NHEADS  16
#define HEADDIM 64
#define CONVDIM 1056         // DINNER + 2*DSTATE
#define DINPROJ 2096         // 2*DINNER + 2*DSTATE + NHEADS
#define EPSV    1e-5f

// ---------------- fp32 -> bf16 convert ----------------
__global__ __launch_bounds__(256) void cvt_bf16(const float* __restrict__ in,
                                                __bf16* __restrict__ out,
                                                long long n) {
    long long i = (long long)blockIdx.x * 256 + threadIdx.x;
    if (i < n) out[i] = (__bf16)in[i];
}

// convert x -> bf16, also write the L-flipped copy (TDM needs contiguous rows)
__global__ __launch_bounds__(256) void cvt_x_bf16(const float* __restrict__ in,
                                                  __bf16* __restrict__ out,
                                                  __bf16* __restrict__ outrev) {
    long long i = (long long)blockIdx.x * 256 + threadIdx.x;
    if (i >= MROWS * DMODEL) return;
    long long m = i / DMODEL;
    int c = (int)(i - m * DMODEL);
    int l = (int)(m % LSEQ);
    long long b0 = m - l;
    __bf16 v = (__bf16)in[i];
    out[i] = v;
    outrev[(b0 + (LSEQ - 1 - l)) * DMODEL + c] = v;
}

// ---------------- TDM: 2-D tile load, global -> LDS -------------------------
// Loads `tileRows` rows of 32 bf16 (64 B) each, row stride = strideElems bf16,
// zero-filling rows >= validRows. LDS gets 16 B of pad after every 64 B row
// (pad_interval=3 -> 16 DWORDs, pad_amount=3 -> 4 DWORDs) => 40-element stride.
__device__ __forceinline__ void tdm_load_2d(unsigned ldsAddr, const void* gptr,
                                            unsigned tileRows, unsigned validRows,
                                            unsigned strideElems) {
    unsigned long long ga = (unsigned long long)gptr;
    u32x4 g0;
    g0[0] = 1u;                                               // count=1, user mode
    g0[1] = ldsAddr;                                          // lds_addr
    g0[2] = (unsigned)ga;                                     // global_addr[31:0]
    g0[3] = (unsigned)((ga >> 32) & 0x01FFFFFFu) | (2u << 30);// global_addr[56:32] | type=2
    u32x8 g1;
    g1[0] = (1u << 16) | (1u << 20) | (3u << 22) | (3u << 25);// data_size=2B, pad_en, intv=16dw, amt=4dw
    g1[1] = 32u << 16;                                        // tensor_dim0 = 32 elems
    g1[2] = (validRows & 0xFFFFu) << 16;                      // tensor_dim1 lo
    g1[3] = ((validRows >> 16) & 0xFFFFu) | (32u << 16);      // tensor_dim1 hi | tile_dim0=32
    g1[4] = tileRows & 0xFFFFu;                               // tile_dim1 | tile_dim2=0
    g1[5] = strideElems;                                      // tensor_dim0_stride lo32
    g1[6] = 0u;                                               // stride hi | dim1_stride lo
    g1[7] = 0u;                                               // dim1_stride hi
    asm volatile("tensor_load_to_lds %0, %1" :: "s"(g0), "s"(g1) : "memory");
}

// ---------------- WMMA bf16 GEMM:  C[M,N] = A[M,K] @ W[N,K]^T ----------------
// TDM double-buffered tile pipeline; flipC: store C row (b, L-1-l);
// storeHalf: bf16 store into outH at column offset colOff, else fp32 to outF.
__global__ __launch_bounds__(256) void gemm_bf16_wmma(
    const __bf16* __restrict__ A, const __bf16* __restrict__ Wt,
    float* __restrict__ outF, __bf16* __restrict__ outH,
    int M, int N, int K, int Lseq,
    int flipC, int storeHalf, int colOff, int ldOut)
{
    const int LDT = 40; // bf16 elements per LDS row (32 + 8 pad, via TDM padding)
    __shared__ __align__(16) __bf16 As[2][128 * 40];
    __shared__ __align__(16) __bf16 Bs[2][128 * 40];

    const int tid  = threadIdx.x;
    const int wave = tid >> 5;
    const int lane = tid & 31;
    const int laneLo = lane & 15;
    const int hi8  = (lane >> 4) * 8;   // A-fragment K base
    const int hi16 = (lane >> 4) * 16;  // unused; B fragment uses hi8 too
    const int waveM = (wave & 1) * 64;
    const int waveN = (wave >> 1) * 32;
    const int bm = blockIdx.y * 128;
    const int bn = blockIdx.x * 128;

    const __bf16* Abase = A  + (size_t)bm * K;
    const __bf16* Bbase = Wt + (size_t)bn * K;
    unsigned validB = (unsigned)(((N - bn) < 128) ? (N - bn) : 128);

    unsigned ldsA[2] = { (unsigned)(unsigned long long)(&As[0][0]),
                         (unsigned)(unsigned long long)(&As[1][0]) };
    unsigned ldsB[2] = { (unsigned)(unsigned long long)(&Bs[0][0]),
                         (unsigned)(unsigned long long)(&Bs[1][0]) };

    v8f acc[4][2];
#pragma unroll
    for (int a = 0; a < 4; ++a)
#pragma unroll
        for (int b = 0; b < 2; ++b)
#pragma unroll
            for (int q = 0; q < 8; ++q) acc[a][b][q] = 0.0f;

    const int niter = K >> 5;
    if (wave == 0) {
        tdm_load_2d(ldsA[0], Abase, 128, 128,    (unsigned)K);
        tdm_load_2d(ldsB[0], Bbase, 128, validB, (unsigned)K);
    }

    for (int i = 0; i < niter; ++i) {
        const int cur = i & 1;
        if (wave == 0) __builtin_amdgcn_s_wait_tensorcnt(0);   // current tiles resident
        __syncthreads();                                        // publish to all waves
        if (wave == 0 && (i + 1) < niter) {                     // overlap next tile DMA
            tdm_load_2d(ldsA[cur ^ 1], Abase + (size_t)(i + 1) * 32, 128, 128,    (unsigned)K);
            tdm_load_2d(ldsB[cur ^ 1], Bbase + (size_t)(i + 1) * 32, 128, validB, (unsigned)K);
        }

        // hoist ALL fragment loads so the DS pipeline drains under the WMMA burst
        v16bf bfrag[2];
#pragma unroll
        for (int fn = 0; fn < 2; ++fn) {
            int col = waveN + fn * 16 + laneLo;
            const __bf16* bp = &Bs[cur][col * LDT + hi8];
            v8bf lo = *(const v8bf*)(bp);
            v8bf hh = *(const v8bf*)(bp + 16);
            bfrag[fn] = __builtin_shufflevector(lo, hh, 0,1,2,3,4,5,6,7,8,9,10,11,12,13,14,15);
        }
        v16bf afrag[4];
#pragma unroll
        for (int fm = 0; fm < 4; ++fm) {
            int row = waveM + fm * 16 + laneLo;
            const __bf16* ap = &As[cur][row * LDT + hi8];
            v8bf lo = *(const v8bf*)(ap);
            v8bf hh = *(const v8bf*)(ap + 16);
            afrag[fm] = __builtin_shufflevector(lo, hh, 0,1,2,3,4,5,6,7,8,9,10,11,12,13,14,15);
        }
#pragma unroll
        for (int fm = 0; fm < 4; ++fm)
#pragma unroll
            for (int fn = 0; fn < 2; ++fn) {
                acc[fm][fn] = __builtin_amdgcn_wmma_f32_16x16x32_bf16(
                    false, afrag[fm], false, bfrag[fn], (short)0, acc[fm][fn], false, false);
            }
        // next iteration's leading barrier guarantees all waves finished these
        // LDS reads before wave 0 issues the TDM that overwrites this buffer
    }

    // epilogue: 32-bit C layout — vgpr i: lanes<16 -> M=i, lanes>=16 -> M=8+i; N = lane&15
#pragma unroll
    for (int fm = 0; fm < 4; ++fm)
#pragma unroll
        for (int fn = 0; fn < 2; ++fn)
#pragma unroll
            for (int i = 0; i < 8; ++i) {
                int r = bm + waveM + fm * 16 + hi8 + i;
                int c = bn + waveN + fn * 16 + laneLo;
                if (c >= N) continue;
                int rr = r;
                if (flipC) { int bb = r / Lseq, ll = r - bb * Lseq; rr = bb * Lseq + (Lseq - 1 - ll); }
                float v = acc[fm][fn][i];
                if (storeHalf) outH[(size_t)rr * ldOut + colOff + c] = (__bf16)v;
                else           outF[(size_t)rr * ldOut + colOff + c] = v;
            }
}

// ---------------- depthwise causal conv1d (K=4) + SiLU ----------------
__global__ __launch_bounds__(256) void conv_silu(const float* __restrict__ zx,
                                                 const float* __restrict__ convw,
                                                 const float* __restrict__ convb,
                                                 float* __restrict__ xconv) {
    long long idx = (long long)blockIdx.x * 256 + threadIdx.x;
    if (idx >= MROWS * CONVDIM) return;
    int c = (int)(idx % CONVDIM);
    long long m = idx / CONVDIM;
    int l = (int)(m % LSEQ);
    long long base = m - l;                 // b*L
    float acc = convb[c];
#pragma unroll
    for (int i = 0; i < 4; ++i) {
        int lt = l - 3 + i;
        if (lt >= 0) acc += zx[(base + lt) * DINPROJ + DINNER + c] * convw[c * 4 + i];
    }
    float sg = 1.0f / (1.0f + __expf(-acc));
    xconv[idx] = acc * sg;
}

// ---------------- dt = softplus(raw + bias), dA = exp(dt * -exp(Alog)) ----------------
__global__ __launch_bounds__(256) void dt_kernel(const float* __restrict__ zx,
                                                 const float* __restrict__ dtbias,
                                                 const float* __restrict__ Alog,
                                                 float* __restrict__ dtA,
                                                 float* __restrict__ dAA) {
    long long idx = (long long)blockIdx.x * 256 + threadIdx.x;
    if (idx >= MROWS * NHEADS) return;
    int h = (int)(idx % NHEADS);
    long long m = idx / NHEADS;
    float raw = zx[m * DINPROJ + DINNER + CONVDIM + h] + dtbias[h];
    float dt = raw > 20.0f ? raw : log1pf(__expf(raw));
    float A = -__expf(Alog[h]);
    dtA[idx] = dt;
    dAA[idx] = __expf(dt * A);
}

// ---------------- recurrent SSM scan: one block per (b, h); thread = (p, n) state ----------------
__global__ __launch_bounds__(1024) void scan_kernel(const float* __restrict__ xconv,
                                                    const float* __restrict__ dtA,
                                                    const float* __restrict__ dAA,
                                                    float* __restrict__ yArr) {
    int bh = blockIdx.x;
    long long b = bh >> 4;
    int h = bh & 15;
    int tid = threadIdx.x;
    int p = tid >> 4;          // 0..63 : lanes 0-15 of each half-wave share p
    int n = tid & 15;          // 0..15 : state dim
    float hst = 0.0f;
    long long mbase = b * LSEQ;
    for (int t = 0; t < LSEQ; ++t) {
        long long m = mbase + t;
        float dt = dtA[m * NHEADS + h];
        float dA = dAA[m * NHEADS + h];
        float xv = xconv[m * CONVDIM + h * HEADDIM + p];
        float Bv = xconv[m * CONVDIM + DINNER + n];
        float Cv = xconv[m * CONVDIM + DINNER + DSTATE + n];
        hst = hst * dA + (dt * Bv) * xv;
        float yp = hst * Cv;
        // reduce over n (16 lanes; xor 1/2/4/8 stays inside each 16-lane group)
        yp += __shfl_xor(yp, 1, 32);
        yp += __shfl_xor(yp, 2, 32);
        yp += __shfl_xor(yp, 4, 32);
        yp += __shfl_xor(yp, 8, 32);
        if (n == 0) yArr[m * DINNER + h * HEADDIM + p] = yp;
    }
}

// ---------------- gate with silu(z), RMSNorm, -> bf16 ----------------
__global__ __launch_bounds__(256) void gate_rms(const float* __restrict__ yArr,
                                                const float* __restrict__ xconv,
                                                const float* __restrict__ zx,
                                                const float* __restrict__ Dp,
                                                const float* __restrict__ normw,
                                                __bf16* __restrict__ ybf) {
    long long m = blockIdx.x;
    int tid = threadIdx.x;
    __shared__ float red[256];
    float vals[4];
    float ss = 0.0f;
#pragma unroll
    for (int j = 0; j < 4; ++j) {
        int c = j * 256 + tid;
        float yv = yArr[m * DINNER + c];
        float xv = xconv[m * CONVDIM + c];
        float zv = zx[m * DINPROJ + c];
        float sg = 1.0f / (1.0f + __expf(-zv));
        float v = (yv + Dp[c >> 6] * xv) * (zv * sg);
        vals[j] = v;
        ss += v * v;
    }
    red[tid] = ss;
    __syncthreads();
    for (int s = 128; s > 0; s >>= 1) {
        if (tid < s) red[tid] += red[tid + s];
        __syncthreads();
    }
    float rms = rsqrtf(red[0] * (1.0f / (float)DINNER) + EPSV);
#pragma unroll
    for (int j = 0; j < 4; ++j) {
        int c = j * 256 + tid;
        ybf[m * DINNER + c] = (__bf16)(vals[j] * rms * normw[c]);
    }
}

// ---------------- residual + bias + LayerNorm -> d_out ----------------
__global__ __launch_bounds__(256) void residual_ln(const float* __restrict__ x,
                                                   const float* __restrict__ xo,
                                                   const float* __restrict__ pb,
                                                   const float* __restrict__ g,
                                                   const float* __restrict__ beta,
                                                   float* __restrict__ out) {
    long long m = blockIdx.x;
    int tid = threadIdx.x;
    __shared__ float rs[256], rq[256];
    int c1 = 256 + tid;
    float v0 = x[m * DMODEL + tid] + xo[m * DMODEL + tid] + pb[tid];
    float v1 = x[m * DMODEL + c1]  + xo[m * DMODEL + c1]  + pb[c1];
    rs[tid] = v0 + v1;
    rq[tid] = v0 * v0 + v1 * v1;
    __syncthreads();
    for (int s = 128; s > 0; s >>= 1) {
        if (tid < s) { rs[tid] += rs[tid + s]; rq[tid] += rq[tid + s]; }
        __syncthreads();
    }
    float mu  = rs[0] * (1.0f / (float)DMODEL);
    float var = rq[0] * (1.0f / (float)DMODEL) - mu * mu;
    float inv = rsqrtf(var + EPSV);
    out[m * DMODEL + tid] = (v0 - mu) * inv * g[tid] + beta[tid];
    out[m * DMODEL + c1]  = (v1 - mu) * inv * g[c1]  + beta[c1];
}

// ---------------- host driver ----------------
extern "C" void kernel_launch(void* const* d_in, const int* in_sizes, int n_in,
                              void* d_out, int out_size, void* d_ws, size_t ws_size,
                              hipStream_t stream) {
    (void)in_sizes; (void)n_in; (void)out_size; (void)ws_size;
    const float* x      = (const float*)d_in[0];
    const float* Win[2]   = {(const float*)d_in[1],  (const float*)d_in[9]};
    const float* convw[2] = {(const float*)d_in[2],  (const float*)d_in[10]};
    const float* convb[2] = {(const float*)d_in[3],  (const float*)d_in[11]};
    const float* dtb[2]   = {(const float*)d_in[4],  (const float*)d_in[12]};
    const float* Alog[2]  = {(const float*)d_in[5],  (const float*)d_in[13]};
    const float* Dp[2]    = {(const float*)d_in[6],  (const float*)d_in[14]};
    const float* nw[2]    = {(const float*)d_in[7],  (const float*)d_in[15]};
    const float* Wout[2]  = {(const float*)d_in[8],  (const float*)d_in[16]};
    const float* projW = (const float*)d_in[17];
    const float* projb = (const float*)d_in[18];
    const float* lng   = (const float*)d_in[19];
    const float* lnb   = (const float*)d_in[20];
    float* outp = (float*)d_out;

    char* base = (char*)d_ws;
    size_t off = 0;
    auto take = [&](size_t bytes) -> char* {
        char* p = base + off;
        off = (off + bytes + 255) & ~(size_t)255;
        return p;
    };
    const size_t M = (size_t)MROWS;
    __bf16* xbf       = (__bf16*)take(M * DMODEL * 2);
    __bf16* xbfrev    = (__bf16*)take(M * DMODEL * 2);
    __bf16* winbf[2]  = {(__bf16*)take((size_t)DINPROJ * DMODEL * 2),
                         (__bf16*)take((size_t)DINPROJ * DMODEL * 2)};
    __bf16* woutbf[2] = {(__bf16*)take((size_t)DMODEL * DINNER * 2),
                         (__bf16*)take((size_t)DMODEL * DINNER * 2)};
    __bf16* projbf    = (__bf16*)take((size_t)DMODEL * DINNER * 2);
    float*  zx        = (float*)take(M * DINPROJ * 4);
    float*  xconv     = (float*)take(M * CONVDIM * 4);
    float*  dtA       = (float*)take(M * NHEADS * 4);
    float*  dAA       = (float*)take(M * NHEADS * 4);
    float*  yArr      = (float*)take(M * DINNER * 4);
    __bf16* ybf       = (__bf16*)take(M * DINNER * 2);
    __bf16* cat       = (__bf16*)take(M * DINNER * 2);
    float*  xo        = (float*)take(M * DMODEL * 4);

    auto cvt = [&](const float* in, __bf16* out, long long n) {
        int g = (int)((n + 255) / 256);
        cvt_bf16<<<g, 256, 0, stream>>>(in, out, n);
    };
    {
        long long nx = (long long)M * DMODEL;
        cvt_x_bf16<<<(int)((nx + 255) / 256), 256, 0, stream>>>(x, xbf, xbfrev);
    }
    cvt(Win[0], winbf[0], (long long)DINPROJ * DMODEL);
    cvt(Win[1], winbf[1], (long long)DINPROJ * DMODEL);
    cvt(Wout[0], woutbf[0], (long long)DMODEL * DINNER);
    cvt(Wout[1], woutbf[1], (long long)DMODEL * DINNER);
    cvt(projW, projbf, (long long)DMODEL * DINNER);

    dim3 gIn((DINPROJ + 127) / 128, (int)(M / 128));   // 17 x 120
    dim3 gOut((DMODEL + 127) / 128, (int)(M / 128));   // 4 x 120

    for (int d = 0; d < 2; ++d) {
        // in-projection: zx = x(dir) @ Win^T   (bwd uses pre-flipped copy)
        gemm_bf16_wmma<<<gIn, 256, 0, stream>>>(
            d ? xbfrev : xbf, winbf[d], zx, nullptr,
            (int)M, DINPROJ, DMODEL, LSEQ,
            /*flipC=*/0, /*storeHalf=*/0, 0, DINPROJ);

        long long nconv = (long long)M * CONVDIM;
        conv_silu<<<(int)((nconv + 255) / 256), 256, 0, stream>>>(zx, convw[d], convb[d], xconv);

        long long ndt = (long long)M * NHEADS;
        dt_kernel<<<(int)((ndt + 255) / 256), 256, 0, stream>>>(zx, dtb[d], Alog[d], dtA, dAA);

        scan_kernel<<<16 * NHEADS, 1024, 0, stream>>>(xconv, dtA, dAA, yArr);

        gate_rms<<<(int)M, 256, 0, stream>>>(yArr, xconv, zx, Dp[d], nw[d], ybf);

        // out-projection into concat buffer (bf16), un-flip rows for bwd
        gemm_bf16_wmma<<<gOut, 256, 0, stream>>>(
            ybf, woutbf[d], nullptr, cat,
            (int)M, DMODEL, DINNER, LSEQ,
            /*flipC=*/d, /*storeHalf=*/1, /*colOff=*/d * DMODEL, DINNER);
    }

    // final projection: xo = cat @ projW^T
    gemm_bf16_wmma<<<gOut, 256, 0, stream>>>(
        cat, projbf, xo, nullptr,
        (int)M, DMODEL, DINNER, LSEQ, 0, 0, 0, DMODEL);

    residual_ln<<<(int)M, 256, 0, stream>>>(x, xo, projb, lng, lnb, outp);
}
